// SupUniformLoss_66640712565307
// MI455X (gfx1250) — compile-verified, matrix operand order checked by reading, not run on hardware
//
#include <hip/hip_runtime.h>
#include <cstdint>
#include <cstddef>

// ---------------------------------------------------------------------------
// Problem constants (match reference)
// ---------------------------------------------------------------------------
#define N_CLS     10000
#define FEAT_DIM  512
#define BATCH     8192
#define GROUPS    625        // N_CLS / 16 (exact)
#define KCHUNKS   16         // FEAT_DIM / 32 (exact)
#define INV_TEMP  10.0f      // 1 / TEMP
#define EMA_M     0.99f
#define EMA_1M    (1.0f - 0.99f)
#define NORM_EPS  1e-12f

typedef __attribute__((ext_vector_type(16))) __bf16       v16bf;
typedef __attribute__((ext_vector_type(8)))  float        v8f;
typedef __attribute__((ext_vector_type(4)))  float        v4f;
typedef __attribute__((ext_vector_type(8)))  unsigned int v8u;

// RNE f32 -> bf16 (top 16 bits)
static __device__ __forceinline__ unsigned int bf16_rne(float x) {
    unsigned int u = __float_as_uint(x);
    u += 0x7FFFu + ((u >> 16) & 1u);
    return (u >> 16) & 0xFFFFu;
}
static __device__ __forceinline__ float bf16_hi_to_f(unsigned int h16) {
    return __uint_as_float(h16 << 16);
}

// ---------------------------------------------------------------------------
// Kernel 1: per-class sequential EMA + renormalize. One wave32 per class.
// Lane t owns elements [t*16, t*16+16) of the 512-float row.
// ---------------------------------------------------------------------------
extern "C" __global__ __launch_bounds__(256)
void k_ema(const float* __restrict__ feats, const int* __restrict__ labels,
           const float* __restrict__ protos_in, float* __restrict__ protos_out) {
    const int lane = threadIdx.x & 31;
    const int cls  = blockIdx.x * 8 + (threadIdx.x >> 5);   // 1250*8 == 10000 exactly

    float p[16];
    {
        const v4f* src = (const v4f*)(protos_in + (size_t)cls * FEAT_DIM + lane * 16);
        #pragma unroll
        for (int q = 0; q < 4; ++q) {
            v4f v = src[q];
            p[4*q+0] = v.x; p[4*q+1] = v.y; p[4*q+2] = v.z; p[4*q+3] = v.w;
        }
    }

    for (int j0 = 0; j0 < BATCH; j0 += 32) {
        const int lab = labels[j0 + lane];
        unsigned long long m = __ballot(lab == cls);        // wave32: low 32 bits
        while (m) {                                          // ascending sample order
            const int b = __ffsll(m) - 1;
            m &= (m - 1);
            const v4f* f = (const v4f*)(feats + (size_t)(j0 + b) * FEAT_DIM + lane * 16);
            float ss = 0.0f;
            #pragma unroll
            for (int q = 0; q < 4; ++q) {
                v4f v = f[q];
                float e0 = p[4*q+0] * EMA_M + v.x * EMA_1M;
                float e1 = p[4*q+1] * EMA_M + v.y * EMA_1M;
                float e2 = p[4*q+2] * EMA_M + v.z * EMA_1M;
                float e3 = p[4*q+3] * EMA_M + v.w * EMA_1M;
                p[4*q+0] = e0; p[4*q+1] = e1; p[4*q+2] = e2; p[4*q+3] = e3;
                ss += e0*e0 + e1*e1 + e2*e2 + e3*e3;
            }
            #pragma unroll
            for (int off = 16; off >= 1; off >>= 1) ss += __shfl_xor(ss, off, 32);
            const float scale = 1.0f / fmaxf(sqrtf(ss), NORM_EPS);
            #pragma unroll
            for (int e = 0; e < 16; ++e) p[e] *= scale;
        }
    }

    v4f* dst = (v4f*)(protos_out + (size_t)cls * FEAT_DIM + lane * 16);
    #pragma unroll
    for (int q = 0; q < 4; ++q) {
        v4f v; v.x = p[4*q+0]; v.y = p[4*q+1]; v.z = p[4*q+2]; v.w = p[4*q+3];
        dst[q] = v;
    }
}

// ---------------------------------------------------------------------------
// Kernel 2: pack f32 protos into WMMA bf16 A/B register layout, hi/lo split.
// Per (group g, k-chunk c): 32 lanes x 8 dwords (16 bf16 = one wave operand).
// ISA 16-bit A layout: lane L -> row m=L%16, K-half h=L/16;
//   vgpr v in [0,3]: K = c*32 + h*8 + 2v{+0,1}; v in [4,7]: K = c*32+16 + h*8 + 2(v-4){+0,1}.
// Stored lane-contiguous (8 dwords = 32 B per lane) for b128 loads in the GEMM.
// ---------------------------------------------------------------------------
extern "C" __global__ __launch_bounds__(256)
void k_pack(const float* __restrict__ protos, unsigned int* __restrict__ hiBuf,
            unsigned int* __restrict__ loBuf) {
    const int lane = threadIdx.x & 31;
    const int pid  = blockIdx.x * 8 + (threadIdx.x >> 5);   // 0..9999 == g*16 + c
    const int g = pid >> 4, c = pid & 15;
    const int mrow = lane & 15, half = lane >> 4;
    const float* row = protos + (size_t)(g * 16 + mrow) * FEAT_DIM;

    float x[16];
    const v4f* r1 = (const v4f*)(row + c * 32 + half * 8);        // K run for v=0..3
    const v4f* r2 = (const v4f*)(row + c * 32 + 16 + half * 8);   // K run for v=4..7
    v4f a0 = r1[0], a1 = r1[1], b0 = r2[0], b1 = r2[1];
    x[0]=a0.x; x[1]=a0.y; x[2]=a0.z; x[3]=a0.w; x[4]=a1.x; x[5]=a1.y; x[6]=a1.z; x[7]=a1.w;
    x[8]=b0.x; x[9]=b0.y; x[10]=b0.z; x[11]=b0.w; x[12]=b1.x; x[13]=b1.y; x[14]=b1.z; x[15]=b1.w;

    v8u h, l;
    #pragma unroll
    for (int v = 0; v < 8; ++v) {
        float e0 = x[2*v], e1 = x[2*v+1];
        unsigned int h0 = bf16_rne(e0), h1 = bf16_rne(e1);
        float r0  = e0 - bf16_hi_to_f(h0);
        float r1f = e1 - bf16_hi_to_f(h1);
        h[v] = h0 | (h1 << 16);
        l[v] = bf16_rne(r0) | (bf16_rne(r1f) << 16);
    }
    const size_t base = ((size_t)pid * 32 + lane) * 8;
    *(v8u*)(hiBuf + base) = h;
    *(v8u*)(loBuf + base) = l;
}

extern "C" __global__ void k_zero(float* __restrict__ p, int n) {
    int i = blockIdx.x * blockDim.x + threadIdx.x;
    if (i < n) p[i] = 0.0f;
}

// ---------------------------------------------------------------------------
// Kernel 3: symmetric G = P P^T. Only upper-triangle tiles (gj >= gi) are
// computed (halves WMMA *and* exp work). For gj > gi the exp'd tile feeds:
//   rows gi*16+m  via per-lane register row accumulators (summed over n), and
//   rows gj*16+n  via tile column sums (exp(g_ij) == exp(g_ji)).
// Diagonal tiles (gj == gi) contribute rows only, with the diagonal zeroed.
// Anti-diagonal row pairing: block b handles gi = b and gi = 624-b, whose
// upper-triangle tile counts sum to a constant 626 -> balanced grid of 313
// blocks x 16 waves; wave w strides gj by 16.
// Per K-chunk: 4x(2x b128) loads, 3 chained v_wmma_f32_16x16x32_bf16
// (hi*hi + hi*lo + lo*hi, shared f32 accumulator ~ f32-accurate dot).
// ---------------------------------------------------------------------------
extern "C" __global__ __launch_bounds__(512)
void k_gemm_sym(const unsigned int* __restrict__ hiBuf, const unsigned int* __restrict__ loBuf,
                float* __restrict__ rowsum) {
    const int lane = threadIdx.x & 31;
    const int wave = threadIdx.x >> 5;     // 0..15
    const int b    = blockIdx.x;           // 0..312
    const int ncol = lane & 15;            // C/D layout: N = lane%16
    const int half = lane >> 4;
    const int moff = half * 8;             // rows r for lanes 0-15, r+8 for 16-31

    #pragma unroll 1
    for (int sel = 0; sel < 2; ++sel) {
        const int gi = sel ? (GROUPS - 1 - b) : b;
        if (sel && gi == b) break;         // center row (b == 312) done once

        float racc[8];
        #pragma unroll
        for (int r = 0; r < 8; ++r) racc[r] = 0.0f;

        for (int gj = gi + wave; gj < GROUPS; gj += 16) {
            v8f acc = {};
            #pragma unroll 2
            for (int c = 0; c < KCHUNKS; ++c) {
                const size_t abase = (((size_t)(gi * 16 + c)) * 32 + lane) * 8;
                const size_t bbase = (((size_t)(gj * 16 + c)) * 32 + lane) * 8;
                v16bf ah = __builtin_bit_cast(v16bf, *(const v8u*)(hiBuf + abase));
                v16bf al = __builtin_bit_cast(v16bf, *(const v8u*)(loBuf + abase));
                v16bf bh = __builtin_bit_cast(v16bf, *(const v8u*)(hiBuf + bbase));
                v16bf bl = __builtin_bit_cast(v16bf, *(const v8u*)(loBuf + bbase));
                acc = __builtin_amdgcn_wmma_f32_16x16x32_bf16(false, ah, false, bh, (short)0, acc, false, false);
                acc = __builtin_amdgcn_wmma_f32_16x16x32_bf16(false, ah, false, bl, (short)0, acc, false, false);
                acc = __builtin_amdgcn_wmma_f32_16x16x32_bf16(false, al, false, bh, (short)0, acc, false, false);
            }
            const bool diag = (gj == gi);
            float csum = 0.0f;
            #pragma unroll
            for (int r = 0; r < 8; ++r) {
                float e = __expf(acc[r] * INV_TEMP);
                if (diag && (r + moff) == ncol) e = 0.0f;   // zero the true diagonal
                racc[r] += e;                                // row partial (over n)
                csum    += e;                                // column partial (over m)
            }
            if (!diag) {                                     // mirror tile -> rows gj*16+n
                csum += __shfl_xor(csum, 16, 32);            // join the two m-halves
                if (half == 0)
                    atomicAdd(rowsum + gj * 16 + ncol, csum);
            }
        }

        // flush row partials: sum the 16 column-lanes of each row
        #pragma unroll
        for (int r = 0; r < 8; ++r) {
            float s = racc[r];
            #pragma unroll
            for (int off = 1; off < 16; off <<= 1) s += __shfl_xor(s, off, 32);
            if (ncol == 0)                                   // lanes 0 and 16
                atomicAdd(rowsum + gi * 16 + moff + r, s);
        }
    }
}

// ---------------------------------------------------------------------------
// Kernel 4: loss = mean(log(rowsum_i)) - log(N-1)   (TEMP/BASE_TEMP == 1)
// ---------------------------------------------------------------------------
extern "C" __global__ __launch_bounds__(256)
void k_finalize(const float* __restrict__ rowsum, float* __restrict__ out) {
    __shared__ double sh[256];
    double s = 0.0;
    for (int i = threadIdx.x; i < N_CLS; i += 256) s += (double)logf(rowsum[i]);
    sh[threadIdx.x] = s;
    __syncthreads();
    for (int step = 128; step > 0; step >>= 1) {
        if ((int)threadIdx.x < step) sh[threadIdx.x] += sh[threadIdx.x + step];
        __syncthreads();
    }
    if (threadIdx.x == 0)
        out[0] = (float)(sh[0] / (double)N_CLS - log((double)(N_CLS - 1)));
}

// ---------------------------------------------------------------------------
// Launch. Workspace layout (needs ~41 MB):
//   [0, 20480000)            f32 updated prototypes
//   [20480000, 30720000)     packed bf16 "hi"
//   [30720000, 40960000)     packed bf16 "lo"
//   [40960000, 41000000)     f32 rowsum (zero-initialized per call)
// ---------------------------------------------------------------------------
extern "C" void kernel_launch(void* const* d_in, const int* in_sizes, int n_in,
                              void* d_out, int out_size, void* d_ws, size_t ws_size,
                              hipStream_t stream) {
    const float* feats  = (const float*)d_in[0];
    const int*   labels = (const int*)d_in[1];
    const float* protos = (const float*)d_in[2];
    float* out = (float*)d_out;
    char*  ws  = (char*)d_ws;

    float*        protosF = (float*)ws;
    unsigned int* hiBuf   = (unsigned int*)(ws + 20480000);
    unsigned int* loBuf   = (unsigned int*)(ws + 30720000);
    float*        rowsum  = (float*)(ws + 40960000);

    k_ema     <<<GROUPS * 2, 256, 0, stream>>>(feats, labels, protos, protosF); // 1250 blk * 8 waves
    k_pack    <<<GROUPS * 2, 256, 0, stream>>>(protosF, hiBuf, loBuf);
    k_zero    <<<(N_CLS + 255) / 256, 256, 0, stream>>>(rowsum, N_CLS);
    k_gemm_sym<<<(GROUPS + 1) / 2, 512, 0, stream>>>(hiBuf, loBuf, rowsum);     // 313 blocks
    k_finalize<<<1, 256, 0, stream>>>(rowsum, out);
}
